// VectorAttention_19026705121903
// MI455X (gfx1250) — compile-verified, matrix operand order checked by reading, not run on hardware
//
#include <hip/hip_runtime.h>
#include <hip/hip_bf16.h>

#define BB   4
#define CC   32
#define HH   256
#define WW   256
#define HIDN 64
#define HWP  (HH * WW)
#define EPSV 1e-5f

typedef __attribute__((ext_vector_type(16))) __bf16 bf16x16;
typedef __attribute__((ext_vector_type(8)))  float  f32x8;

union BFV {
  bf16x16  v;
  __bf16   e[16];
  unsigned u[8];
  uint4    q4[2];
};

__device__ __forceinline__ int iclamp(int x, int lo, int hi) {
  return x < lo ? lo : (x > hi ? hi : x);
}

static __device__ __forceinline__ f32x8 wmma_bf16(bf16x16 a, bf16x16 b, f32x8 c) {
  // D = A(16x32 bf16) x B(32x16 bf16) + C(16x16 f32)
  return __builtin_amdgcn_wmma_f32_16x16x32_bf16(false, a, false, b, (short)0, c,
                                                 false, false);
}

// ---------------------------------------------------------------------------
// Kernel 1: fused 1x1 convs.  D(128 x Npix) = [Wqkv ; Wg] * x.
//   m-tiles 0..1 -> q, 2..3 -> -k (negated!), 4..5 -> v  (bf16, pixel-major)
//   m-tiles 6..7 -> gate: d_out = gate + bg + x  (f32, B,C,H,W layout)
// Each wave: preloads all 8 A (weight) tiles, loops 8 N-tiles of 16 pixels.
// ---------------------------------------------------------------------------
__global__ __launch_bounds__(256) void va_qkvg(
    const float* __restrict__ x,
    const float* __restrict__ Wqkv,
    const float* __restrict__ Wg,
    const float* __restrict__ bg,
    __bf16* __restrict__ qws,
    __bf16* __restrict__ kws,     // holds -k
    __bf16* __restrict__ vws,
    float* __restrict__ out)
{
  const int lane = threadIdx.x & 31;
  const int wave = threadIdx.x >> 5;
  const int grp  = lane >> 4;      // 0: lanes 0-15 (K lo half), 1: lanes 16-31
  const int lm   = lane & 15;
  const int k0   = grp ? 8 : 0;    // A/B K sub-offset per lane half

  // A tiles: row (out-ch) per lane, 16 K values packed 2/VGPR.
  bf16x16 A[8];
  #pragma unroll
  for (int m = 0; m < 8; ++m) {
    const int o = m * 16 + lm;
    const float* wrow = (o < 96) ? (Wqkv + o * CC) : (Wg + (o - 96) * CC);
    BFV av;
    #pragma unroll
    for (int i = 0; i < 8; ++i) av.e[i]     = (__bf16)wrow[k0 + i];
    #pragma unroll
    for (int i = 0; i < 8; ++i) av.e[8 + i] = (__bf16)wrow[16 + k0 + i];
    A[m] = av.v;
  }

  float bgl[16];
  #pragma unroll
  for (int j = 0; j < 2; ++j)
    #pragma unroll
    for (int r = 0; r < 8; ++r)
      bgl[j * 8 + r] = bg[j * 16 + r + 8 * grp];

  const long pix0 = (long)blockIdx.x * 1024 + (long)wave * 128;
  for (int t = 0; t < 8; ++t) {
    const long pixbase = pix0 + (long)t * 16;
    const int  b    = (int)(pixbase / HWP);
    const int  pimg = (int)(pixbase % HWP);

    // B operand: x tile, column (pixel) per lane, K packed (strided loads).
    BFV bv;
    const float* xb = x + ((long)b * CC) * HWP + pimg + lm;
    #pragma unroll
    for (int i = 0; i < 8; ++i) bv.e[i]     = (__bf16)xb[(long)(k0 + i) * HWP];
    #pragma unroll
    for (int i = 0; i < 8; ++i) bv.e[8 + i] = (__bf16)xb[(long)(16 + k0 + i) * HWP];

    f32x8 acc[8];
    #pragma unroll
    for (int m = 0; m < 8; ++m) {
      f32x8 cz = {};
      acc[m] = wmma_bf16(A[m], bv.v, cz);
    }

    // D layout: element (lane, r) = (row 16m + r + 8*grp, pixel pixbase + lm)
    const long pix = pixbase + lm;
    __bf16* qp = qws + pix * CC;
    __bf16* kp = kws + pix * CC;
    __bf16* vp = vws + pix * CC;
    #pragma unroll
    for (int j = 0; j < 2; ++j) {
      #pragma unroll
      for (int r = 0; r < 8; ++r) {
        const int ch = j * 16 + r + 8 * grp;
        qp[ch] = (__bf16)acc[0 + j][r];
        kp[ch] = (__bf16)(-acc[2 + j][r]);   // store -k
        vp[ch] = (__bf16)acc[4 + j][r];
      }
    }
    #pragma unroll
    for (int j = 0; j < 2; ++j) {
      #pragma unroll
      for (int r = 0; r < 8; ++r) {
        const int  cb = j * 16 + r + 8 * grp;
        const long oi = ((long)b * CC + cb) * HWP + pimg + lm;
        out[oi] = acc[6 + j][r] + bgl[j * 8 + r] + x[oi];
      }
    }
  }
}

// ---------------------------------------------------------------------------
// Kernel 2: neighbor MLP + softmax + weighted v.  One wave owns 16 pixels
// (fixed h, consecutive w).  GEMM rows = pixels; neighbors are the outer
// loop so the 9-way softmax is a per-element running accumulation.
// ---------------------------------------------------------------------------
__global__ __launch_bounds__(256) void va_attn(
    const __bf16* __restrict__ qws,
    const __bf16* __restrict__ kws,   // -k
    const __bf16* __restrict__ vws,
    const float* __restrict__ W1,
    const float* __restrict__ b1,
    const float* __restrict__ ln_w,
    const float* __restrict__ ln_b,
    const float* __restrict__ W2,
    const float* __restrict__ b2,
    float* __restrict__ out)
{
  __shared__ __bf16 lds_h[8][16 * HIDN];   // per-wave D->A transpose scratch

  const int lane = threadIdx.x & 31;
  const int wave = threadIdx.x >> 5;
  const int grp  = lane >> 4;
  const int lm   = lane & 15;
  const int k0   = grp ? 8 : 0;

  const int tile = blockIdx.x;             // 2048 tiles
  const int tw = tile & 15;                // 16 tiles across W
  const int th = (tile >> 4) & 31;         // 32 tiles across H
  const int b  = tile >> 9;
  const int h  = th * 8 + wave;
  const int w0 = tw * 16;

  // B1: W1^T (32 x 64), 4 N-tiles.  B2: W2^T (64 x 32), 2 N-tiles x 2 K-steps.
  bf16x16 B1t[4];
  #pragma unroll
  for (int n = 0; n < 4; ++n) {
    const float* wr = W1 + (n * 16 + lm) * CC;
    BFV t;
    #pragma unroll
    for (int i = 0; i < 8; ++i) t.e[i]     = (__bf16)wr[k0 + i];
    #pragma unroll
    for (int i = 0; i < 8; ++i) t.e[8 + i] = (__bf16)wr[16 + k0 + i];
    B1t[n] = t.v;
  }
  bf16x16 B2t[2][2];
  #pragma unroll
  for (int c = 0; c < 2; ++c)
    #pragma unroll
    for (int ks = 0; ks < 2; ++ks) {
      const float* wr = W2 + (c * 16 + lm) * HIDN + ks * 32;
      BFV t;
      #pragma unroll
      for (int i = 0; i < 8; ++i) t.e[i]     = (__bf16)wr[k0 + i];
      #pragma unroll
      for (int i = 0; i < 8; ++i) t.e[8 + i] = (__bf16)wr[16 + k0 + i];
      B2t[c][ks] = t.v;
    }

  float b1l[4], lwl[4], lbl[4];
  #pragma unroll
  for (int n = 0; n < 4; ++n) {
    const int d = n * 16 + lm;
    b1l[n] = b1[d]; lwl[n] = ln_w[d]; lbl[n] = ln_b[d];
  }
  float b2l[2];
  #pragma unroll
  for (int c = 0; c < 2; ++c) b2l[c] = b2[c * 16 + lm];

  // q in A layout (row m = lm, pixel w0+lm), kept packed bf16: 2x b128 loads.
  const long prow = ((long)b * HH + h) * WW;
  const uint4* qrow4 = (const uint4*)(qws + (prow + w0 + lm) * CC);
  BFV qv;
  qv.q4[0] = qrow4[grp];
  qv.q4[1] = qrow4[2 + grp];

  f32x8 num[2] = {};
  f32x8 den[2] = {};

  for (int nb = 0; nb < 9; ++nb) {
    const int dy = nb / 3 - 1;
    const int dx = nb % 3 - 1;
    const int hh = h + dy;
    const bool hok = (hh >= 0) && (hh < HH);
    const int hhc = iclamp(hh, 0, HH - 1);
    const long nrow = ((long)b * HH + hhc) * WW;

    __builtin_prefetch(vws + (nrow + iclamp(w0 + dx, 0, WW - 1)) * CC);

    // A = q + (-k)(neighbor); out-of-image k == 0 (zero-padded unfold).
    const int wwA = w0 + lm + dx;
    const bool aok = hok && (wwA >= 0) && (wwA < WW);
    const uint4* krow4 =
        (const uint4*)(kws + (nrow + iclamp(wwA, 0, WW - 1)) * CC);
    BFV kv;
    kv.q4[0] = krow4[grp];
    kv.q4[1] = krow4[2 + grp];
    #pragma unroll
    for (int i = 0; i < 8; ++i) kv.u[i] = aok ? kv.u[i] : 0u;
    BFV av;
    av.v = qv.v + kv.v;                 // packed bf16 add (k pre-negated)

    // hid = A * W1^T + b1 : 16x64 in 4 accumulators.
    f32x8 hacc[4];
    #pragma unroll
    for (int n = 0; n < 4; ++n) {
      f32x8 cz = {};
      hacc[n] = wmma_bf16(av.v, B1t[n], cz);
      #pragma unroll
      for (int r = 0; r < 8; ++r) hacc[n][r] += b1l[n];
    }

    // LayerNorm per row (row = VGPR r within a 16-lane half).
    f32x8 s  = hacc[0] + hacc[1] + hacc[2] + hacc[3];
    f32x8 s2 = hacc[0] * hacc[0] + hacc[1] * hacc[1]
             + hacc[2] * hacc[2] + hacc[3] * hacc[3];
    float mean[8], inv[8];
    #pragma unroll
    for (int r = 0; r < 8; ++r) {
      float sv = s[r], qsum = s2[r];
      #pragma unroll
      for (int mask = 1; mask <= 8; mask <<= 1) {
        sv   += __shfl_xor(sv, mask, 32);
        qsum += __shfl_xor(qsum, mask, 32);
      }
      const float mu = sv * (1.0f / HIDN);
      float var = qsum * (1.0f / HIDN) - mu * mu;
      var = var > 0.f ? var : 0.f;
      mean[r] = mu;
      inv[r] = 1.0f / (__builtin_sqrtf(var) + EPSV);
    }

    // Normalize + ReLU, re-layout D->A via per-wave LDS scratch.
    __bf16* hp = &lds_h[wave][0];
    #pragma unroll
    for (int n = 0; n < 4; ++n)
      #pragma unroll
      for (int r = 0; r < 8; ++r) {
        float hv = (hacc[n][r] - mean[r]) * inv[r];
        hv = lwl[n] * hv + lbl[n];
        hv = hv > 0.f ? hv : 0.f;
        const int m = r + 8 * grp;
        hp[m * HIDN + n * 16 + lm] = (__bf16)hv;
      }
    asm volatile("s_wait_dscnt 0" ::: "memory");

    // A operand of the second GEMM: row = pixel lm, two K-steps of 32.
    const uint4* hrow4 = (const uint4*)(hp + lm * HIDN);
    bf16x16 ah[2];
    #pragma unroll
    for (int ks = 0; ks < 2; ++ks) {
      BFV t;
      t.q4[0] = hrow4[ks * 4 + grp];
      t.q4[1] = hrow4[ks * 4 + 2 + grp];
      ah[ks] = t.v;
    }

    // scores = hid * W2^T + b2 (K=64: two chained WMMAs), then online
    // softmax-weighted accumulation of v.
    #pragma unroll
    for (int c = 0; c < 2; ++c) {
      f32x8 cz = {};
      f32x8 sc = wmma_bf16(ah[0], B2t[c][0], cz);
      sc = wmma_bf16(ah[1], B2t[c][1], sc);
      #pragma unroll
      for (int r = 0; r < 8; ++r) {
        const int m = r + 8 * grp;
        const int wwv = w0 + m + dx;
        const bool vok = hok && (wwv >= 0) && (wwv < WW);
        const float vval = vok
            ? (float)vws[(nrow + iclamp(wwv, 0, WW - 1)) * CC + c * 16 + lm]
            : 0.f;
        const float e = __expf(sc[r] + b2l[c]);
        den[c][r] += e;
        num[c][r] += e * vval;
      }
    }
  }

  // out += weighted_v  (out already holds base + x from kernel 1).
  #pragma unroll
  for (int c = 0; c < 2; ++c)
    #pragma unroll
    for (int r = 0; r < 8; ++r) {
      const int  m  = r + 8 * grp;
      const int  ch = c * 16 + lm;
      const long oi = ((long)b * CC + ch) * HWP + (long)h * WW + (w0 + m);
      out[oi] += num[c][r] / den[c][r];
    }
}

extern "C" void kernel_launch(void* const* d_in, const int* in_sizes, int n_in,
                              void* d_out, int out_size, void* d_ws, size_t ws_size,
                              hipStream_t stream) {
  (void)in_sizes; (void)n_in; (void)out_size; (void)ws_size;
  const float* x    = (const float*)d_in[0];
  const float* Wqkv = (const float*)d_in[1];
  const float* W1   = (const float*)d_in[2];
  const float* b1   = (const float*)d_in[3];
  const float* ln_w = (const float*)d_in[4];
  const float* ln_b = (const float*)d_in[5];
  const float* W2   = (const float*)d_in[6];
  const float* b2   = (const float*)d_in[7];
  const float* Wg   = (const float*)d_in[8];
  const float* bg   = (const float*)d_in[9];
  float* out = (float*)d_out;

  char* ws = (char*)d_ws;
  const size_t npix   = (size_t)BB * HWP;
  const size_t planeB = npix * CC * sizeof(__bf16);   // 16 MiB per bf16 tensor
  __bf16* qws = (__bf16*)(ws);
  __bf16* kws = (__bf16*)(ws + planeB);
  __bf16* vws = (__bf16*)(ws + 2 * planeB);

  // Kernel 1: 256 blocks x 8 waves x 128 pixels = 262144 pixels.
  va_qkvg<<<dim3(256), dim3(256), 0, stream>>>(x, Wqkv, Wg, bg, qws, kws, vws, out);
  // Kernel 2: 2048 tiles (16 w-tiles x 32 h-tiles x 4 batches), 8 waves/tile.
  va_attn<<<dim3(2048), dim3(256), 0, stream>>>(qws, kws, vws, W1, b1, ln_w, ln_b,
                                                W2, b2, out);
}
